// Model_19542101197007
// MI455X (gfx1250) — compile-verified, matrix operand order checked by reading, not run on hardware
//
#include <hip/hip_runtime.h>

// ---------------------------------------------------------------- types ----
typedef __attribute__((ext_vector_type(16))) _Float16 v16h;
typedef __attribute__((ext_vector_type(8)))  _Float16 v8h;
typedef __attribute__((ext_vector_type(8)))  float    v8f;
typedef int v4i __attribute__((vector_size(16)));   // async-LDS builtin operand

#ifndef __has_builtin
#define __has_builtin(x) 0
#endif
#if __has_builtin(__builtin_amdgcn_global_load_async_to_lds_b128) && \
    __has_builtin(__builtin_amdgcn_s_wait_asynccnt)
#define USE_ASYNC_LDS 1
#else
#define USE_ASYNC_LDS 0
#endif

static inline int ceil_div(int a, int b) { return (a + b - 1) / b; }

// ------------------------------------------------------------- kernels ----

__global__ void zero_f32(float* __restrict__ p, int n) {
    int i = blockIdx.x * blockDim.x + threadIdx.x;
    if (i < n) p[i] = 0.f;
}

// front-end: t = relu(feats @ W + b), feats [N,32], W [32,3]
__global__ void front_linear(const float* __restrict__ feats,
                             const float* __restrict__ W,
                             const float* __restrict__ b,
                             float* __restrict__ t, int N) {
    int n = blockIdx.x * blockDim.x + threadIdx.x;
    if (n >= N) return;
    const float* f = feats + (size_t)n * 32;
    for (int q = 0; q < 3; ++q) {
        float s = b[q];
        for (int c = 0; c < 32; ++c) s += f[c] * W[c * 3 + q];
        t[(size_t)n * 3 + q] = fmaxf(s, 0.f);
    }
}

// per-channel sum / sumsq with LDS tree reduction, one atomic per block
__global__ __launch_bounds__(256)
void bn_stats(const float* __restrict__ x, int N, int C,
              float* __restrict__ sum, float* __restrict__ sumsq) {
    __shared__ float s1[256];
    __shared__ float s2[256];
    const int c   = blockIdx.y;
    const int row = blockIdx.x * blockDim.x + threadIdx.x;
    float v = 0.f;
    if (row < N) v = x[(size_t)row * C + c];
    s1[threadIdx.x] = v;
    s2[threadIdx.x] = v * v;
    __syncthreads();
    for (int s = 128; s > 0; s >>= 1) {
        if (threadIdx.x < s) {
            s1[threadIdx.x] += s1[threadIdx.x + s];
            s2[threadIdx.x] += s2[threadIdx.x + s];
        }
        __syncthreads();
    }
    if (threadIdx.x == 0) {
        atomicAdd(&sum[c], s1[0]);
        atomicAdd(&sumsq[c], s2[0]);
    }
}

__global__ void bn_apply(const float* __restrict__ x, float* __restrict__ y,
                         const float* __restrict__ sum, const float* __restrict__ sumsq,
                         const float* __restrict__ g, const float* __restrict__ b,
                         int N, int C, float eps, int relu) {
    size_t i = (size_t)blockIdx.x * blockDim.x + threadIdx.x;
    if (i >= (size_t)N * C) return;
    int c = (int)(i % C);
    float invN = 1.f / (float)N;
    float mu   = sum[c] * invN;
    float var  = sumsq[c] * invN - mu * mu;
    float v = (x[i] - mu) * rsqrtf(var + eps) * g[c] + b[c];
    if (relu) v = fmaxf(v, 0.f);
    y[i] = v;
}

// f32 [N,C] -> f16 [N,Cpad] zero-padded
__global__ void cvt_pad(const float* __restrict__ x, _Float16* __restrict__ y,
                        int N, int C, int Cpad) {
    size_t i = (size_t)blockIdx.x * blockDim.x + threadIdx.x;
    if (i >= (size_t)N * Cpad) return;
    int    c = (int)(i % Cpad);
    size_t n = i / Cpad;
    y[i] = (c < C) ? (_Float16)x[n * C + c] : (_Float16)0.f;
}

// Pack W[k][ci][co] (f32) into WMMA B-operand lane layout, f16:
//   Wp[((k*nct+ct)*ncot+cot)*32 + lane][j] ,  kk = (lane>>4)*8 + j + (j>=8?8:0)
//   ci = ct*32+kk (zero-pad past Cin), co = cot*16 + (lane&15)
__global__ void pack_w(const float* __restrict__ W, _Float16* __restrict__ Wp,
                       int K, int Cin, int Cout, int Cpad) {
    const int nct  = Cpad >> 5;
    const int ncot = Cout >> 4;
    size_t total = (size_t)K * nct * ncot * 512;
    size_t i = (size_t)blockIdx.x * blockDim.x + threadIdx.x;
    if (i >= total) return;
    int j    = (int)(i & 15);
    int lane = (int)((i >> 4) & 31);
    size_t r = i >> 9;
    int cot = (int)(r % ncot); r /= ncot;
    int ct  = (int)(r % nct);
    int k   = (int)(r / nct);
    int kk = ((lane >> 4) << 3) + j + ((j >= 8) ? 8 : 0);
    int ci = ct * 32 + kk;
    int co = cot * 16 + (lane & 15);
    Wp[i] = (ci < Cin) ? (_Float16)W[((size_t)k * Cin + ci) * Cout + co]
                       : (_Float16)0.f;
}

// stage one packed-weight k-slice (slice halfs, 16B-chunked) into LDS
__device__ __forceinline__
void stage_slice(const _Float16* __restrict__ g, _Float16* __restrict__ l,
                 int slice, int tid) {
    const int chunks = slice >> 3;                 // 16-byte chunks
    for (int c = tid; c < chunks; c += 256) {
#if USE_ASYNC_LDS
        __builtin_amdgcn_global_load_async_to_lds_b128(
            (v4i*)(const_cast<_Float16*>(g) + (size_t)c * 8),
            (v4i*)(l + (size_t)c * 8), 0, 0);
#else
        *(v8h*)(l + (size_t)c * 8) = *(const v8h*)(g + (size_t)c * 8);
#endif
    }
}

__device__ __forceinline__ void wait_async_lds() {
#if USE_ASYNC_LDS
    __builtin_amdgcn_s_wait_asynccnt(0);
#endif
}

// Submanifold 3x3x3 gather-conv: one wave per 16-row x 16-cout tile.
// Weights double-buffered through LDS (async global->LDS per k-tap slice);
// acc(16x16,f32) += A(16x32,f16 gathered rows) x B(32x16,f16 from LDS).
__global__ __launch_bounds__(256)
void subconv_wmma(const _Float16* __restrict__ xh,   // [N, Cpad] f16
                  const int* __restrict__ nbr,        // [N, 27]
                  const _Float16* __restrict__ whp,   // packed weights
                  float* __restrict__ out,            // [N, Cout] f32
                  int N, int Cpad, int Cout) {
    extern __shared__ _Float16 ldsw[];               // 2 * slice halfs
    const int nct   = Cpad >> 5;
    const int ncot  = Cout >> 4;
    const int slice = nct * ncot * 512;              // halfs per k-slice
    const int tid   = threadIdx.x;
    const int wave  = tid >> 5;
    const int lane  = tid & 31;
    const int nrt   = (N + 15) >> 4;
    const int ntile = nrt * ncot;

    int tile = blockIdx.x * 8 + wave;
    const bool live = (tile < ntile);
    if (!live) tile = 0;                 // clamp: keep wave in barriers/WMMA
    const int rt  = tile / ncot;
    const int cot = tile % ncot;
    const int lr  = lane & 15;
    const int hi  = lane >> 4;
    const int row = rt * 16 + lr;
    const bool row_ok = live && (row < N);
    const int kk_lo = hi * 8;

    stage_slice(whp, ldsw, slice, tid);              // k = 0 -> buffer 0

    const _Float16 z = (_Float16)0.f;
    v8f acc = {0.f, 0.f, 0.f, 0.f, 0.f, 0.f, 0.f, 0.f};

    for (int k = 0; k < 27; ++k) {
        wait_async_lds();
        __syncthreads();                             // slice k ready
        if (k + 1 < 27)
            stage_slice(whp + (size_t)(k + 1) * slice,
                        ldsw + (size_t)((k + 1) & 1) * slice, slice, tid);

        int n = -1;
        if (row_ok) n = nbr[(size_t)row * 27 + k];
        const _Float16* bbase = ldsw + (size_t)(k & 1) * slice;

        for (int ct = 0; ct < nct; ++ct) {
            v8h alo = {z, z, z, z, z, z, z, z};
            v8h ahi = {z, z, z, z, z, z, z, z};
            if (n >= 0) {
                const _Float16* ap = xh + (size_t)n * Cpad + ct * 32 + kk_lo;
                alo = *(const v8h*)(ap);             // K = kk_lo .. +7
                ahi = *(const v8h*)(ap + 16);        // K = kk_lo+16 .. +23
            }
            v16h a = __builtin_shufflevector(alo, ahi,
                0, 1, 2, 3, 4, 5, 6, 7, 8, 9, 10, 11, 12, 13, 14, 15);
            const _Float16* bp =
                bbase + ((size_t)(ct * ncot + cot) * 32 + lane) * 16;
            v16h b = *(const v16h*)bp;               // ds_load from LDS
            acc = __builtin_amdgcn_wmma_f32_16x16x32_f16(
                false, a, false, b, (short)0, acc, false, false);
        }
        __syncthreads();                             // done reading slice k
    }

    if (live) {
        const int co = cot * 16 + lr;
#pragma unroll
        for (int j = 0; j < 8; ++j) {
            int rm = rt * 16 + hi * 8 + j;           // D: VGPR j -> M=j(+8)
            if (rm < N) out[(size_t)rm * Cout + co] = acc[j];
        }
    }
}

// stride-2 conv: contrib = y[n,:] @ Wd[off[n]] ; scatter-add to parent (segment_sum)
__global__ void down_conv(const float* __restrict__ y, const float* __restrict__ Wd,
                          const int* __restrict__ off, const int* __restrict__ par,
                          float* __restrict__ out, int N, int Cin, int Cout) {
    size_t i = (size_t)blockIdx.x * blockDim.x + threadIdx.x;
    if (i >= (size_t)N * Cout) return;
    int co = (int)(i % Cout);
    int n  = (int)(i / Cout);
    const float* w  = Wd + ((size_t)off[n] * Cin) * Cout + co;
    const float* yy = y + (size_t)n * Cin;
    float s = 0.f;
    for (int c = 0; c < Cin; ++c) s += yy[c] * w[(size_t)c * Cout];
    atomicAdd(&out[(size_t)par[n] * Cout + co], s);
}

// stride-2 deconv gather + JoinTable: cat[:, :C]=skip, cat[:, C:]=up
__global__ void up_conv_cat(const float* __restrict__ ycoarse,
                            const float* __restrict__ skip,
                            const float* __restrict__ Wu,
                            const int* __restrict__ off, const int* __restrict__ par,
                            float* __restrict__ cat, int N, int Cin, int Cout) {
    size_t i = (size_t)blockIdx.x * blockDim.x + threadIdx.x;
    if (i >= (size_t)N * Cout) return;
    int co = (int)(i % Cout);
    int n  = (int)(i / Cout);
    const float* w  = Wu + ((size_t)off[n] * Cin) * Cout + co;
    const float* yy = ycoarse + (size_t)par[n] * Cin;
    float s = 0.f;
    for (int c = 0; c < Cin; ++c) s += yy[c] * w[(size_t)c * Cout];
    cat[(size_t)n * (2 * Cout) + Cout + co] = s;
    cat[(size_t)n * (2 * Cout) + co]        = skip[i];
}

// classifier head: out = x @ Wlin + blin, x [N,16], Wlin [16,2]
__global__ void head(const float* __restrict__ x, const float* __restrict__ Wlin,
                     const float* __restrict__ blin, float* __restrict__ out, int N) {
    int n = blockIdx.x * blockDim.x + threadIdx.x;
    if (n >= N) return;
    for (int q = 0; q < 2; ++q) {
        float s = blin[q];
        for (int c = 0; c < 16; ++c) s += x[(size_t)n * 16 + c] * Wlin[c * 2 + q];
        out[(size_t)n * 2 + q] = s;
    }
}

// ---------------------------------------------------------------- host ----

extern "C" void kernel_launch(void* const* d_in, const int* in_sizes, int n_in,
                              void* d_out, int out_size, void* d_ws, size_t ws_size,
                              hipStream_t stream) {
    (void)out_size; (void)ws_size; (void)n_in;
    static const int PLANES[7] = {16, 32, 48, 64, 80, 96, 112};

    // ---- input map (pytree-leaf order of setup_inputs dict) ----
    int p = 0;
    const int*   coords     = (const int*)d_in[p++];  (void)coords;
    const float* feats      = (const float*)d_in[p++];
    const float* compress_W = (const float*)d_in[p++];
    const float* compress_b = (const float*)d_in[p++];
    const float* g0v        = (const float*)d_in[p++];
    const float* b0v        = (const float*)d_in[p++];
    const float* W_in       = (const float*)d_in[p++];
    const float *gA[7], *bA[7], *WA[7];
    for (int l = 0; l < 7; ++l) gA[l] = (const float*)d_in[p++];
    for (int l = 0; l < 7; ++l) bA[l] = (const float*)d_in[p++];
    for (int l = 0; l < 7; ++l) WA[l] = (const float*)d_in[p++];
    const float *gD[6], *bD[6], *Wd[6], *gU[6], *bU[6], *Wu[6], *gB[6], *bB[6], *WB[6];
    for (int l = 0; l < 6; ++l) gD[l] = (const float*)d_in[p++];
    for (int l = 0; l < 6; ++l) bD[l] = (const float*)d_in[p++];
    for (int l = 0; l < 6; ++l) Wd[l] = (const float*)d_in[p++];
    for (int l = 0; l < 6; ++l) gU[l] = (const float*)d_in[p++];
    for (int l = 0; l < 6; ++l) bU[l] = (const float*)d_in[p++];
    for (int l = 0; l < 6; ++l) Wu[l] = (const float*)d_in[p++];
    for (int l = 0; l < 6; ++l) gB[l] = (const float*)d_in[p++];
    for (int l = 0; l < 6; ++l) bB[l] = (const float*)d_in[p++];
    for (int l = 0; l < 6; ++l) WB[l] = (const float*)d_in[p++];
    const float* gF   = (const float*)d_in[p++];
    const float* bF   = (const float*)d_in[p++];
    const float* Wlin = (const float*)d_in[p++];
    const float* blin = (const float*)d_in[p++];
    const int* nbrs[7];
    for (int l = 0; l < 7; ++l) nbrs[l] = (const int*)d_in[p++];
    const int *pars[6], *offs[6];
    for (int l = 0; l < 6; ++l) pars[l] = (const int*)d_in[p++];
    for (int l = 0; l < 6; ++l) offs[l] = (const int*)d_in[p++];

    int NL[7];
    for (int l = 0; l < 7; ++l) NL[l] = in_sizes[86 + l] / 27;
    const int N0 = NL[0];

    // ---- workspace bump allocator ----
    char* ws = (char*)d_ws;
    size_t woff = 0;
    auto alloc = [&](size_t bytes) -> void* {
        void* q = ws + woff;
        woff = (woff + bytes + 255) & ~(size_t)255;
        return q;
    };
    float* sumb = (float*)alloc(512 * sizeof(float));
    float* sqb  = sumb + 256;
    float* t    = (float*)alloc((size_t)N0 * 3 * sizeof(float));
    float* G1   = (float*)alloc((size_t)N0 * 32 * sizeof(float));
    float* G2   = (float*)alloc((size_t)N0 * 32 * sizeof(float));
    float* G3   = (float*)alloc((size_t)N0 * 32 * sizeof(float));
    _Float16* bh = (_Float16*)alloc((size_t)N0 * 32 * sizeof(_Float16));
    _Float16* wh = (_Float16*)alloc((size_t)27 * 224 * 128 * sizeof(_Float16));
    float* enc_x[7];
    for (int l = 0; l < 7; ++l)
        enc_x[l] = (float*)alloc((size_t)NL[l] * PLANES[l] * sizeof(float));

    // ---- launch helpers ----
    auto bn = [&](const float* x, float* y, const float* g, const float* b,
                  int N, int C, float eps, int relu) {
        zero_f32<<<2, 256, 0, stream>>>(sumb, 512);
        bn_stats<<<dim3(ceil_div(N, 256), C), 256, 0, stream>>>(x, N, C, sumb, sqb);
        bn_apply<<<ceil_div(N * C, 256), 256, 0, stream>>>(x, y, sumb, sqb, g, b,
                                                           N, C, eps, relu);
    };
    auto subconv = [&](const float* x, const int* nbr, const float* W,
                       float* y, int N, int Cin, int Cout) {
        int Cpad = (Cin + 31) & ~31;
        cvt_pad<<<ceil_div(N * Cpad, 256), 256, 0, stream>>>(x, bh, N, Cin, Cpad);
        int wtot = 27 * Cpad * Cout;
        pack_w<<<ceil_div(wtot, 256), 256, 0, stream>>>(W, wh, 27, Cin, Cout, Cpad);
        int nwaves  = ceil_div(N, 16) * (Cout / 16);
        int nblocks = ceil_div(nwaves, 8);          // 8 waves / 256-thread block
        int shbytes = 4 * Cpad * Cout;              // 2 double-buffered k-slices
        subconv_wmma<<<nblocks, 256, shbytes, stream>>>(bh, nbr, wh, y, N, Cpad, Cout);
    };

    // ---- front-end: Linear -> ReLU -> BatchNorm1d(eps 1e-5) -> SubConv 3->16 ----
    front_linear<<<ceil_div(N0, 256), 256, 0, stream>>>(feats, compress_W, compress_b, t, N0);
    bn(t, t, g0v, b0v, N0, 3, 1e-5f, 0);
    subconv(t, nbrs[0], W_in, G1, N0, 3, 16);
    float* x = G1;

    // ---- UNet encoder ----
    for (int l = 0; l < 7; ++l) {
        const int C = PLANES[l];
        bn(x, G3, gA[l], bA[l], NL[l], C, 1e-4f, 1);
        subconv(G3, nbrs[l], WA[l], enc_x[l], NL[l], C, C);
        x = enc_x[l];
        if (l < 6) {
            const int Cn = PLANES[l + 1];
            bn(x, G3, gD[l], bD[l], NL[l], C, 1e-4f, 1);
            zero_f32<<<ceil_div(NL[l + 1] * Cn, 256), 256, 0, stream>>>(G1, NL[l + 1] * Cn);
            down_conv<<<ceil_div(NL[l] * Cn, 256), 256, 0, stream>>>(
                G3, Wd[l], offs[l], pars[l], G1, NL[l], C, Cn);
            x = G1;
        }
    }

    // ---- UNet decoder ----
    for (int l = 5; l >= 0; --l) {
        const int C = PLANES[l], Cn = PLANES[l + 1];
        bn(x, G1, gU[l], bU[l], NL[l + 1], Cn, 1e-4f, 1);
        up_conv_cat<<<ceil_div(NL[l] * C, 256), 256, 0, stream>>>(
            G1, enc_x[l], Wu[l], offs[l], pars[l], G2, NL[l], Cn, C);
        bn(G2, G3, gB[l], bB[l], NL[l], 2 * C, 1e-4f, 1);
        subconv(G3, nbrs[l], WB[l], G1, NL[l], 2 * C, C);
        x = G1;
    }

    // ---- final BNReLU(16) + classifier ----
    bn(x, G3, gF, bF, N0, 16, 1e-4f, 1);
    head<<<ceil_div(N0, 256), 256, 0, stream>>>(G3, Wlin, blin, (float*)d_out, N0);
}
